// KPDetector_36034775613846
// MI455X (gfx1250) — compile-verified, matrix operand order checked by reading, not run on hardware
//
#include <hip/hip_runtime.h>
#include <hip/hip_bf16.h>

// ---------------------------------------------------------------------------
// Problem constants (from reference):
//   feature_map (8,32,16,64,64) f32 ; contact head K=15 ch ; force head 90 ch
//   conv3d 3x3x3 pad=1  ->  implicit GEMM  M=B*D*H*W=524288, N=105, K=32*27=864
//   softmax over DHW (T=0.1), soft-argmax vs grid, heatmap-weighted force sum
// ---------------------------------------------------------------------------

#define B_    8
#define C_    32
#define D_    16
#define H_    64
#define W_    64
#define K_    15
#define NCH   105          // 15 contact + 90 force output channels
#define NPAD  112          // padded to 7 tiles of 16
#define NTILE 7
#define KDIM  864          // 27 taps * 32 channels
#define DHW   (D_*H_*W_)   // 65536 voxels per batch
#define NVOX  (B_*DHW)     // 524288
#define INV_T 10.0f        // 1 / temperature
#define CHUNKS_PER_BK 64   // reduction blocks per (b,k)

#define WTILE_ELEMS (16 * KDIM)              // bf16 elems per weight N-tile
#define WTILE_BYTES (WTILE_ELEMS * 2)        // 27648 B (contiguous in wbuf)

typedef __attribute__((ext_vector_type(16))) __bf16 v16bf;
typedef __attribute__((ext_vector_type(8)))  float  v8f;
typedef __attribute__((ext_vector_type(4)))  unsigned int v4u;
typedef __attribute__((ext_vector_type(8)))  int v8i;
typedef __attribute__((ext_vector_type(4)))  int v4i;

#if __has_builtin(__builtin_amdgcn_tensor_load_to_lds) && \
    __has_builtin(__builtin_amdgcn_s_wait_tensorcnt)
#define HAVE_TDM 1
#else
#define HAVE_TDM 0
#endif

union ABfrag { v16bf v; uint4 u[2]; };

static __device__ __forceinline__ unsigned short f32_to_bf16(float f) {
    unsigned int u = __float_as_uint(f);
    u += 0x7FFFu + ((u >> 16) & 1u);      // round-to-nearest-even
    return (unsigned short)(u >> 16);
}

#if HAVE_TDM
// Issue a 1-D TDM copy: `units` 8-byte elements, global -> LDS.
// D# layout per CDNA5 ISA:
//   g0: [1:0]=count=1, [63:32]=lds_addr, [120:64]=global_addr, [127:126]=type=2
//   g1: [17:16]=data_size=3 (8B); tensor_dim0=tile_dim0=units; tensor_dim1=1;
//       tile_dim1=0 (1-D); tensor_dim0_stride=units; groups 2/3 zero (<=2D).
// This toolchain exposes the 6-arg builtin:
//   (uint32x4 g0, int32x8 g1, int32x4, int32x4, int32x8, i32 cpol)
static __device__ __forceinline__ void tdm_load_1d(const void* lds_dst,
                                                   const void* global_src,
                                                   unsigned units) {
    // Flat LDS addresses carry the LDS offset in addr[31:0] (ISA aperture rule).
    unsigned lds_off = (unsigned)(unsigned long long)(uintptr_t)lds_dst;
    unsigned long long ga = (unsigned long long)(uintptr_t)global_src;
    v4u g0;
    g0.x = 1u;                                             // count=1 (valid)
    g0.y = lds_off;                                        // lds_addr
    g0.z = (unsigned)(ga & 0xFFFFFFFFull);                 // global_addr[31:0]
    g0.w = (unsigned)((ga >> 32) & 0x01FFFFFFull) | (2u << 30);  // addr[56:32], type=2
    v8i g1;
    g1[0] = (int)(3u << 16);                               // data_size = 3 (8 bytes)
    g1[1] = (int)((units & 0xFFFFu) << 16);                // tensor_dim0[15:0]
    g1[2] = (int)((units >> 16) | (1u << 16));             // tensor_dim0[31:16], tensor_dim1=1
    g1[3] = (int)((units & 0xFFFFu) << 16);                // tile_dim0 = units
    g1[4] = 0;                                             // tile_dim1=0 (1-D), tile_dim2=0
    g1[5] = (int)units;                                    // tensor_dim0_stride[31:0]
    g1[6] = 0;
    g1[7] = 0;
    v4i z4 = {0, 0, 0, 0};
    v8i z8 = {0, 0, 0, 0, 0, 0, 0, 0};
    __builtin_amdgcn_tensor_load_to_lds(g0, g1, z4, z4, z8, 0);
}
#endif

// ---------------------------------------------------------------------------
// Kernel 0a: feature map f32 (b,c,d,h,w)  ->  bf16 (b,d,h,w,c)  (c innermost)
// ---------------------------------------------------------------------------
__global__ __launch_bounds__(256)
void cvt_fmap_kernel(const float* __restrict__ fmap, unsigned short* __restrict__ fmapT) {
    size_t i = (size_t)blockIdx.x * 256 + threadIdx.x;     // 16,777,216 elems
    if (i >= (size_t)NVOX * C_) return;
    int c    = (int)(i & (C_ - 1));
    size_t v = i >> 5;                                      // (b,d,h,w)
    int w = (int)(v & 63);
    int h = (int)((v >> 6) & 63);
    int d = (int)((v >> 12) & 15);
    int b = (int)(v >> 16);
    float f = fmap[((((size_t)b * C_ + c) * D_ + d) * H_ + h) * W_ + w];
    fmapT[i] = f32_to_bf16(f);
}

// ---------------------------------------------------------------------------
// Kernel 0b: pack both weight heads -> bf16 wbuf[n][tap*32+c], n padded to 112;
//            fused bias vector (105 valid, padded with 0).
// ---------------------------------------------------------------------------
__global__ __launch_bounds__(256)
void cvt_wts_kernel(const float* __restrict__ cw, const float* __restrict__ cb,
                    const float* __restrict__ fw, const float* __restrict__ fb,
                    unsigned short* __restrict__ wbuf, float* __restrict__ biasbuf) {
    int i = blockIdx.x * 256 + threadIdx.x;                 // NPAD*KDIM = 96768
    if (i >= NPAD * KDIM) return;
    int n   = i / KDIM;
    int kk  = i - n * KDIM;
    int tap = kk >> 5;                                      // (kd*3+kh)*3+kw
    int c   = kk & 31;
    float val = 0.0f;
    if (n < K_)            val = cw[(size_t)n * KDIM + (size_t)c * 27 + tap];
    else if (n < NCH)      val = fw[(size_t)(n - K_) * KDIM + (size_t)c * 27 + tap];
    wbuf[(size_t)n * KDIM + kk] = f32_to_bf16(val);
    if (i < NPAD) biasbuf[i] = (i < K_) ? cb[i] : (i < NCH ? fb[i - K_] : 0.0f);
}

// ---------------------------------------------------------------------------
// Kernel 1: implicit-GEMM conv3d via v_wmma_f32_16x16x32_bf16.
//   grid = B*D*H blocks (one W-row each), 128 threads = 4 waves.
//   Each wave owns a 16-voxel M-tile; loops 7 N-tiles x 27 taps of WMMA.
//   LDS: input halo (3x3x66 rows x 32ch bf16) + double-buffered weight stage
//        filled by the Tensor Data Mover (tensor_load_to_lds, TENSORcnt),
//        prefetching tile j+1 while tile j is consumed by WMMA.
//   Output: conv[voxel][ch], voxel-major f32, ch stride NCH, + bias.
// ---------------------------------------------------------------------------
__global__ __launch_bounds__(128)
void conv_wmma_kernel(const unsigned short* __restrict__ fmapT,   // (b,d,h,w,c) bf16
                      const unsigned short* __restrict__ wbuf,    // (112, 864) bf16
                      const float* __restrict__ biasbuf,          // (112) f32
                      float* __restrict__ out) {                  // (NVOX, 105) f32
    __shared__ __align__(16) unsigned short xs[9 * 66 * C_];      // ((dz*3+dy)*66+xx)*32+c
    __shared__ __align__(16) unsigned short wls[2][WTILE_ELEMS];  // (nl, tap*32+c) x2

    const int tid  = threadIdx.x;
    const int lane = tid & 31;
    const int wave = tid >> 5;
    const int blk  = blockIdx.x;          // b*1024 + d*64 + h
    const int h = blk & 63;
    const int d = (blk >> 6) & 15;
    const int b = blk >> 10;

#if HAVE_TDM
    // Kick off the DMA of weight N-tile 0 before halo staging (independent).
    if (wave == 0)
        tdm_load_1d(&wls[0][0], wbuf, WTILE_BYTES >> 3);
#endif

    // ---- stage input halo: 9*66 rows of 32 bf16 (64B) = 4 uint4 per row ----
    for (int e = tid; e < 594 * 4; e += 128) {
        int row  = e >> 2;
        int q    = e & 3;
        int dzdy = row / 66;
        int xx   = row - dzdy * 66;
        int dz = dzdy / 3, dy = dzdy - dz * 3;
        int ds = d + dz - 1, hs = h + dy - 1, wsrc = xx - 1;
        uint4 val = make_uint4(0u, 0u, 0u, 0u);
        if ((unsigned)ds < (unsigned)D_ && (unsigned)hs < (unsigned)H_ &&
            (unsigned)wsrc < (unsigned)W_) {
            const uint4* src = reinterpret_cast<const uint4*>(
                fmapT + ((((size_t)b * D_ + ds) * H_ + hs) * W_ + wsrc) * C_);
            val = src[q];
        }
        reinterpret_cast<uint4*>(xs)[e] = val;
    }

    const int m    = lane & 15;     // A: row M ; B: col N ; D: col N
    const int hi   = lane >> 4;     // lane half selects K sub-groups
    const int cofs = hi ? 8 : 0;    // lanes 0-15: K{0..7,16..23}; 16-31: K{8..15,24..31}
    const int w0   = wave * 16;

    for (int j = 0; j < NTILE; ++j) {
#if HAVE_TDM
        if (wave == 0) __builtin_amdgcn_s_wait_tensorcnt(0);  // tile j landed
        __syncthreads();                                      // release all waves
        if (wave == 0 && j + 1 < NTILE)                       // prefetch tile j+1
            tdm_load_1d(&wls[(j + 1) & 1][0],
                        wbuf + (size_t)(j + 1) * WTILE_ELEMS, WTILE_BYTES >> 3);
        const unsigned short* wcur = &wls[j & 1][0];
#else
        __syncthreads();
        for (int e = tid; e < 16 * 108; e += 128) {           // synchronous stage
            int nl = e / 108;
            int q  = e - nl * 108;
            reinterpret_cast<uint4*>(&wls[0][0])[nl * 108 + q] =
                reinterpret_cast<const uint4*>(wbuf + (size_t)(j * 16 + nl) * KDIM)[q];
        }
        __syncthreads();
        const unsigned short* wcur = &wls[0][0];
#endif

        v8f acc = {0.f, 0.f, 0.f, 0.f, 0.f, 0.f, 0.f, 0.f};
        #pragma unroll
        for (int tap = 0; tap < 27; ++tap) {
            const int dz  = tap / 9;
            const int rem = tap - dz * 9;
            const int dy  = rem / 3;
            const int kw  = rem - dy * 3;
            const int xx  = w0 + m + kw;                      // w + kw - 1, pre-shifted
            const int arow = (dz * 3 + dy) * 66 + xx;

            ABfrag a, bb;
            const uint4* ap = reinterpret_cast<const uint4*>(xs + arow * C_ + cofs);
            a.u[0] = ap[0];               // channels cofs..cofs+7
            a.u[1] = ap[2];               // channels cofs+16..cofs+23
            const uint4* bp = reinterpret_cast<const uint4*>(wcur + m * KDIM + tap * 32 + cofs);
            bb.u[0] = bp[0];
            bb.u[1] = bp[2];

            acc = __builtin_amdgcn_wmma_f32_16x16x32_bf16(
                false, a.v, false, bb.v, (short)0, acc, false, false);
        }

        // ---- store D: lane l, vgpr r -> (M = r + 8*(l>>4), N = l&15) ----
        const int N = j * 16 + m;
        if (N < NCH) {
            const float bias = biasbuf[N];
            const size_t base = (size_t)(blk * W_ + w0) * NCH + N;
            #pragma unroll
            for (int r = 0; r < 8; ++r) {
                int M = r + 8 * hi;
                out[base + (size_t)M * NCH] = acc[r] + bias;
            }
        }
    }
}

// ---------------------------------------------------------------------------
// Kernel 2: fused online softmax + soft-argmax + force reduction (partials).
//   One block per (b, k, chunk of 1024 voxels); single pass over conv output.
// ---------------------------------------------------------------------------
__global__ __launch_bounds__(256)
void reduce_partial_kernel(const float* __restrict__ conv, float* __restrict__ partials) {
    const int blk   = blockIdx.x;            // (b*15+k)*64 + chunk
    const int chunk = blk & (CHUNKS_PER_BK - 1);
    const int bk    = blk >> 6;
    const int k     = bk % K_;
    const int b     = bk / K_;
    const int tid   = threadIdx.x;

    float m = -3.4e38f, s = 0.0f;
    float acc[9];
    #pragma unroll
    for (int c = 0; c < 9; ++c) acc[c] = 0.0f;

    #pragma unroll
    for (int i = 0; i < 4; ++i) {
        const int v = chunk * 1024 + i * 256 + tid;          // voxel within batch
        const int dd = v >> 12;
        const int hh = (v >> 6) & 63;
        const int ww = v & 63;
        const float* row = conv + (size_t)(b * DHW + v) * NCH;

        const float logit = row[k] * INV_T;
        const float nm    = fmaxf(m, logit);
        const float scale = __expf(m - nm);
        const float wgt   = __expf(logit - nm);
        s = s * scale + wgt;

        const float crd[3] = { -1.0f + ww * (2.0f / 63.0f),
                               -1.0f + hh * (2.0f / 63.0f),
                               -1.0f + dd * (2.0f / 15.0f) };
        #pragma unroll
        for (int c = 0; c < 3; ++c) acc[c] = acc[c] * scale + wgt * crd[c];
        const float* f = row + K_ + 6 * k;
        #pragma unroll
        for (int c = 0; c < 6; ++c) acc[3 + c] = acc[3 + c] * scale + wgt * f[c];
        m = nm;
    }

    __shared__ float red[256][11];
    red[tid][0] = m;
    red[tid][1] = s;
    #pragma unroll
    for (int c = 0; c < 9; ++c) red[tid][2 + c] = acc[c];
    __syncthreads();

    for (int off = 128; off > 0; off >>= 1) {
        if (tid < off) {
            const float m1 = red[tid][0], m2 = red[tid + off][0];
            const float M  = fmaxf(m1, m2);
            const float e1 = __expf(m1 - M), e2 = __expf(m2 - M);
            red[tid][0] = M;
            #pragma unroll
            for (int c = 1; c < 11; ++c)
                red[tid][c] = red[tid][c] * e1 + red[tid + off][c] * e2;
        }
        __syncthreads();
    }
    if (tid == 0) {
        float* p = partials + (size_t)blk * 11;
        #pragma unroll
        for (int c = 0; c < 11; ++c) p[c] = red[0][c];
    }
}

// ---------------------------------------------------------------------------
// Kernel 3: combine 64 partials per (b,k), normalize, write outputs.
//   d_out[0..359]    = contact (B,K,3)
//   d_out[360..1079] = force   (B,K,2,3) flat
// ---------------------------------------------------------------------------
__global__ __launch_bounds__(128)
void finalize_kernel(const float* __restrict__ partials, float* __restrict__ out) {
    const int t = blockIdx.x * blockDim.x + threadIdx.x;     // t = b*15+k
    if (t >= B_ * K_) return;
    const float* base = partials + (size_t)t * CHUNKS_PER_BK * 11;

    float M = -3.4e38f;
    for (int i = 0; i < CHUNKS_PER_BK; ++i) M = fmaxf(M, base[i * 11]);
    float S = 0.0f, A[9];
    #pragma unroll
    for (int c = 0; c < 9; ++c) A[c] = 0.0f;
    for (int i = 0; i < CHUNKS_PER_BK; ++i) {
        const float* p = base + i * 11;
        const float e = __expf(p[0] - M);
        S += p[1] * e;
        #pragma unroll
        for (int c = 0; c < 9; ++c) A[c] += p[2 + c] * e;
    }
    const float inv = 1.0f / S;
    out[t * 3 + 0] = A[0] * inv;
    out[t * 3 + 1] = A[1] * inv;
    out[t * 3 + 2] = A[2] * inv;
    #pragma unroll
    for (int c = 0; c < 6; ++c) out[B_ * K_ * 3 + t * 6 + c] = A[3 + c] * inv;
}

// ---------------------------------------------------------------------------
extern "C" void kernel_launch(void* const* d_in, const int* in_sizes, int n_in,
                              void* d_out, int out_size, void* d_ws, size_t ws_size,
                              hipStream_t stream) {
    (void)in_sizes; (void)n_in; (void)out_size; (void)ws_size;
    const float* fmap = (const float*)d_in[0];
    const float* cw   = (const float*)d_in[1];
    const float* cb   = (const float*)d_in[2];
    const float* fw   = (const float*)d_in[3];
    const float* fb   = (const float*)d_in[4];

    char* ws = (char*)d_ws;
    auto align256 = [](size_t x) { return (x + 255) & ~(size_t)255; };
    size_t o_fmapT = 0;
    size_t o_wbuf  = align256(o_fmapT + (size_t)NVOX * C_ * sizeof(unsigned short));
    size_t o_bias  = align256(o_wbuf + (size_t)NPAD * KDIM * sizeof(unsigned short));
    size_t o_conv  = align256(o_bias + (size_t)NPAD * sizeof(float));
    size_t o_part  = align256(o_conv + (size_t)NVOX * NCH * sizeof(float));

    unsigned short* fmapT = (unsigned short*)(ws + o_fmapT);
    unsigned short* wbuf  = (unsigned short*)(ws + o_wbuf);
    float*          biasb = (float*)(ws + o_bias);
    float*          conv  = (float*)(ws + o_conv);
    float*          part  = (float*)(ws + o_part);

    cvt_fmap_kernel<<<(NVOX * C_ + 255) / 256, 256, 0, stream>>>(fmap, fmapT);
    cvt_wts_kernel<<<(NPAD * KDIM + 255) / 256, 256, 0, stream>>>(cw, cb, fw, fb, wbuf, biasb);
    conv_wmma_kernel<<<B_ * D_ * H_, 128, 0, stream>>>(fmapT, wbuf, biasb, conv);
    reduce_partial_kernel<<<B_ * K_ * CHUNKS_PER_BK, 256, 0, stream>>>(conv, part);
    finalize_kernel<<<1, 128, 0, stream>>>(part, (float*)d_out);
}
